// LinearSelfAttention_21191368638896
// MI455X (gfx1250) — matrix operand order repaired and checked by reading, not compiled, not run on hardware
//
#include <hip/hip_runtime.h>
#include <hip/hip_bf16.h>
#include <stdint.h>

// ---------------------------------------------------------------------------
// Gated Linear Attention (chunked) forward for MI455X / gfx1250.
// FP32 gating math + FP32 LDS-resident state (with bf16 transposed shadow);
// BF16 WMMA (v_wmma_f32_16x16x32_bf16) for all four chunk GEMMs.
// wave32, 8 waves / block, Vd split across 2 blocks per (b,h).
// Gate tile staged via async global->LDS loads when the toolchain has them.
// ---------------------------------------------------------------------------

typedef __attribute__((ext_vector_type(16))) __bf16 v16bf;
typedef __attribute__((ext_vector_type(8)))  float  v8f;
typedef int v4i_ __attribute__((vector_size(16)));   // pointee type of async b128 builtin

#define T_DIM 4096
#define H_DIM 16
#define K_DIM 128
#define V_DIM 128
#define CH    64
#define NC    (T_DIM / CH)
#define VSPLIT 2
#define VL    (V_DIM / VSPLIT)   // 64 local V columns per block

#if defined(__has_builtin)
#if __has_builtin(__builtin_amdgcn_global_load_async_to_lds_b128) && \
    __has_builtin(__builtin_amdgcn_s_wait_asynccnt)
#define GLA_ASYNC 1
#endif
#endif

#define GAS1(p) ((__attribute__((address_space(1))) v4i_*)(p))
#define LAS3(p) ((__attribute__((address_space(3))) v4i_*)(p))

union FragU { uint4 q[2]; v16bf v; };

// float -> bf16 bits, round-to-nearest-even
__device__ __forceinline__ unsigned f2bf(float f) {
  unsigned u = __builtin_bit_cast(unsigned, f);
  u += 0x7FFFu + ((u >> 16) & 1u);
  return u >> 16;
}

// 16-bit A-matrix 16x32 fragment (ISA 7.12.2): lanes 0-15 K={0..7,16..23},
// lanes 16-31 K={8..15,24..31}.
__device__ __forceinline__ v16bf frag_a(const unsigned short* row, int kk, int hi) {
  const unsigned short* p = row + kk + (hi << 3);
  FragU f;
  f.q[0] = *reinterpret_cast<const uint4*>(p);        // K = hi*8 + 0..7
  f.q[1] = *reinterpret_cast<const uint4*>(p + 16);   // K = hi*8 + 16..23
  return f.v;
}

// 16-bit B-matrix 32x16 fragment: lane holds N=lane&15, K order same as frag_a.
__device__ __forceinline__ v16bf frag_b(const unsigned short* row, int kk, int hi) {
  const unsigned short* p = row + kk + (hi << 3);
  FragU f;
  f.q[0] = *reinterpret_cast<const uint4*>(p);
  f.q[1] = *reinterpret_cast<const uint4*>(p + 16);
  return f.v;
}

__device__ __forceinline__ v8f wmma_bf16(v16bf a, v16bf b, v8f c) {
  return __builtin_amdgcn_wmma_f32_16x16x32_bf16(
      /*neg_a=*/false, a, /*neg_b=*/false, b,
      /*c_mod=*/(short)0, c, /*reuse_a=*/false, /*reuse_b=*/false);
}

__global__ __launch_bounds__(256, 1) void gla_fwd_kernel(
    const float* __restrict__ Q, const float* __restrict__ V,
    const float* __restrict__ G, const float* __restrict__ S0,
    float* __restrict__ Out) {
  extern __shared__ __align__(16) unsigned char smem_raw[];
  unsigned short* sQe  = (unsigned short*)smem_raw;   // [t][k] 64x128 bf16 (q*exp(b)*scale)
  unsigned short* sKin = sQe  + CH * K_DIM;           // [t][k] 64x128 bf16 (q*exp(-b))
  unsigned short* sKoT = sKin + CH * K_DIM;           // [k][t] 128x64 bf16 (q*exp(blast-b), transposed)
  unsigned short* sVt  = sKoT + K_DIM * CH;           // [v][t] 64x64  bf16 (V transposed)
  unsigned short* sA   = sVt  + VL * CH;              // [i][j] 64x64  bf16 (masked A)
  unsigned short* sSbT = sA   + CH * CH;              // [v][k] 64x128 bf16 shadow of state (transposed)
  float* sS   = (float*)(sSbT + VL * K_DIM);          // [k][v] 128x64 f32 resident state
  float* sT   = sS + K_DIM * VL;                      // [t][k] 64x128 f32 gate/cumsum, later o stage
  float* sDec = sT + CH * K_DIM;                      // [k] 128 f32 exp(b_last)
  float* sAux = sDec + K_DIM;                         // [k] 128 f32 cumsum carry

  const int tid  = threadIdx.x;
  const int lane = tid & 31;          // wave32 on gfx1250
  const int wv   = tid >> 5;          // 8 waves
  const int hi   = (lane >> 4) & 1;
  const int l15  = lane & 15;

  const int vs = blockIdx.x & (VSPLIT - 1);
  const int h  = (blockIdx.x / VSPLIT) & (H_DIM - 1);
  const int b0 = blockIdx.x / (VSPLIT * H_DIM);
  const float scale = 0.08838834764831845f;           // 1/sqrt(128)

  // initial state slab (fp32) + bf16 transposed shadow
  for (int idx = tid; idx < K_DIM * VL; idx += 256) {
    int k = idx / VL, v = idx - k * VL;
    float s = S0[(((size_t)b0 * H_DIM + h) * K_DIM + k) * V_DIM + vs * VL + v];
    sS[idx] = s;
    sSbT[v * K_DIM + k] = (unsigned short)f2bf(s);
  }
  __syncthreads();

#pragma unroll 1
  for (int c = 0; c < NC; ++c) {
    const size_t rowbase = ((size_t)b0 * T_DIM + (size_t)c * CH) * H_DIM + h;

    // ---- stage raw gate chunk into sT (async mem->LDS if available) ----
#ifdef GLA_ASYNC
    for (int idx = tid; idx < CH * (K_DIM / 4); idx += 256) {
      int t = idx >> 5, seg = idx & 31;   // 16B segments of each 512B row
      const float* g = G + (rowbase + (size_t)t * H_DIM) * K_DIM + seg * 4;
      __builtin_amdgcn_global_load_async_to_lds_b128(
          GAS1(g), LAS3(sT + t * K_DIM + seg * 4), 0, 0);
    }
    __builtin_amdgcn_s_wait_asynccnt(0);
#else
    for (int idx = tid; idx < CH * K_DIM; idx += 256) {
      int t = idx >> 7, k = idx & (K_DIM - 1);
      sT[idx] = G[(rowbase + (size_t)t * H_DIM) * K_DIM + k];
    }
#endif
    // prefetch next chunk's Q (global_prefetch_b8; one cacheline per thread)
    if (c + 1 < NC) {
      const float* nq = Q + (rowbase + (size_t)CH * H_DIM) * K_DIM;
      int t = tid >> 2, seg = tid & 3;
      __builtin_prefetch(nq + (size_t)t * H_DIM * K_DIM + seg * 32, 0, 0);
    }
    __syncthreads();

    // ---- b = cumsum(-g) along time, 2 threads per k-column ----
    {
      const int col  = tid & (K_DIM - 1);
      const int half = tid >> 7;          // 0: t in [0,32), 1: t in [32,64)
      const int tb   = half * (CH / 2);
      float acc = 0.f;
#pragma unroll 4
      for (int t = 0; t < CH / 2; ++t) {
        acc -= sT[(tb + t) * K_DIM + col];
        sT[(tb + t) * K_DIM + col] = acc;
      }
      if (!half) sAux[col] = acc;
      __syncthreads();
      if (half) {
        float off = sAux[col];
#pragma unroll 4
        for (int t = 0; t < CH / 2; ++t)
          sT[(tb + t) * K_DIM + col] += off;
        sDec[col] = __expf(acc + off);    // exp(b_last)
      }
    }
    __syncthreads();

    // ---- build bf16 operands, 2 adjacent k per thread (gating in fp32) ----
    // qe = q*exp(b)*scale ; kin = q*exp(-b) ; kout = q*exp(blast-b) = q*sDec[k]*exp(-b)
    for (int idx2 = tid; idx2 < CH * (K_DIM / 2); idx2 += 256) {
      int t = idx2 >> 6, k2 = (idx2 & 63) * 2;
      const float* qp = Q + (rowbase + (size_t)t * H_DIM) * K_DIM + k2;
      float q0 = qp[0], q1 = qp[1];                 // k == q in reference
      int base = t * K_DIM + k2;
      float b0v = sT[base], b1v = sT[base + 1];
      float e10 = __expf(b0v),  e11 = __expf(b1v);
      float e20 = __expf(-b0v), e21 = __expf(-b1v);
      unsigned pe = f2bf(q0 * e10 * scale) | (f2bf(q1 * e11 * scale) << 16);
      unsigned pi = f2bf(q0 * e20)         | (f2bf(q1 * e21) << 16);
      *reinterpret_cast<unsigned*>(sQe  + base) = pe;   // base is even
      *reinterpret_cast<unsigned*>(sKin + base) = pi;
      sKoT[k2 * CH + t]       = (unsigned short)f2bf(q0 * (sDec[k2]     * e20));
      sKoT[(k2 + 1) * CH + t] = (unsigned short)f2bf(q1 * (sDec[k2 + 1] * e21));
    }
    for (int idx2 = tid; idx2 < CH * (VL / 2); idx2 += 256) {
      int t = idx2 / (VL / 2), v2 = (idx2 % (VL / 2)) * 2;
      const float* vp = V + (rowbase + (size_t)t * H_DIM) * V_DIM + vs * VL + v2;
      float v0 = vp[0], v1 = vp[1];
      sVt[v2 * CH + t]       = (unsigned short)f2bf(v0);
      sVt[(v2 + 1) * CH + t] = (unsigned short)f2bf(v1);
    }
    __syncthreads();

    // ---- A = tril(q_e @ k_in^T), 16 tiles over 64x64, 2 per wave ----
#pragma unroll
    for (int i = 0; i < 2; ++i) {
      int tile = wv * 2 + i;
      int tm = tile >> 2, tn = tile & 3;
      v8f acc = {};
      if (tm >= tn) {                    // tiles strictly above diagonal are all-zero
        v16bf fa[4], fb[4];
#pragma unroll
        for (int s = 0; s < 4; ++s) {    // preload all k-steps, then MMA chain
          fa[s] = frag_a(sQe  + (tm * 16 + l15) * K_DIM, s * 32, hi);
          fb[s] = frag_b(sKin + (tn * 16 + l15) * K_DIM, s * 32, hi);
        }
#pragma unroll
        for (int s = 0; s < 4; ++s) acc = wmma_bf16(fa[s], fb[s], acc);
      }
#pragma unroll
      for (int r = 0; r < 8; ++r) {      // C layout: M = r + hi*8, N = l15
        int irow = tm * 16 + r + hi * 8;
        int jcol = tn * 16 + l15;
        sA[irow * CH + jcol] = (unsigned short)f2bf((jcol <= irow) ? acc[r] : 0.0f);
      }
    }
    __syncthreads();

    // ---- o = A@v + q_e@S (fused accumulators); kv = koT@v ----
    v8f oacc[2];
    v8f kvacc[4];
#pragma unroll
    for (int i = 0; i < 2; ++i) {
      int tile = wv * 2 + i;
      int tm = tile >> 2, tn = tile & 3;
      v8f acc = {};
      {   // intra-chunk: Kdim = time j (64)
        v16bf fa[2], fb[2];
#pragma unroll
        for (int s = 0; s < 2; ++s) {
          fa[s] = frag_a(sA  + (tm * 16 + l15) * CH, s * 32, hi);
          fb[s] = frag_b(sVt + (tn * 16 + l15) * CH, s * 32, hi);
        }
#pragma unroll
        for (int s = 0; s < 2; ++s) acc = wmma_bf16(fa[s], fb[s], acc);
      }
      {   // inter-chunk: Kdim = k (128), B from bf16 transposed state shadow
        v16bf fa[4], fb[4];
#pragma unroll
        for (int s = 0; s < 4; ++s) {
          fa[s] = frag_a(sQe  + (tm * 16 + l15) * K_DIM, s * 32, hi);
          fb[s] = frag_b(sSbT + (tn * 16 + l15) * K_DIM, s * 32, hi);
        }
#pragma unroll
        for (int s = 0; s < 4; ++s) acc = wmma_bf16(fa[s], fb[s], acc);
      }
      oacc[i] = acc;
    }
#pragma unroll
    for (int i = 0; i < 4; ++i) {        // 32 kv tiles over 128x64, 4 per wave
      int tile = wv * 4 + i;
      int tm = tile >> 2, tn = tile & 3;
      v8f acc = {};
      v16bf fa[2], fb[2];
#pragma unroll
      for (int s = 0; s < 2; ++s) {
        fa[s] = frag_a(sKoT + (tm * 16 + l15) * CH, s * 32, hi);
        fb[s] = frag_b(sVt  + (tn * 16 + l15) * CH, s * 32, hi);
      }
#pragma unroll
      for (int s = 0; s < 2; ++s) acc = wmma_bf16(fa[s], fb[s], acc);
      kvacc[i] = acc;
    }
    __syncthreads();   // all reads of sS/sSbT/sT complete before updates below

    // ---- stage o into sT[0:64*64], decay-update state (fp32 + bf16 shadow) ----
    float* sO = sT;
#pragma unroll
    for (int i = 0; i < 2; ++i) {
      int tile = wv * 2 + i;
      int tm = tile >> 2, tn = tile & 3;
#pragma unroll
      for (int r = 0; r < 8; ++r)
        sO[(tm * 16 + r + hi * 8) * CH + tn * 16 + l15] = oacc[i][r];
    }
#pragma unroll
    for (int i = 0; i < 4; ++i) {
      int tile = wv * 4 + i;
      int tm = tile >> 2, tn = tile & 3;
#pragma unroll
      for (int r = 0; r < 8; ++r) {
        int kg = tm * 16 + r + hi * 8;
        int vg = tn * 16 + l15;
        float ns = sDec[kg] * sS[kg * VL + vg] + kvacc[i][r];
        sS[kg * VL + vg] = ns;
        sSbT[vg * K_DIM + kg] = (unsigned short)f2bf(ns);
      }
    }
    __syncthreads();

    // ---- coalesced store of o ----
    for (int idx = tid; idx < CH * VL; idx += 256) {
      int t = idx / VL, v = idx - t * VL;
      Out[(rowbase + (size_t)t * H_DIM) * V_DIM + vs * VL + v] = sO[idx];
    }
    __syncthreads();   // protect sT/sS/sSbT before next chunk
  }
}

extern "C" void kernel_launch(void* const* d_in, const int* in_sizes, int n_in,
                              void* d_out, int out_size, void* d_ws, size_t ws_size,
                              hipStream_t stream) {
  const float* Q  = (const float*)d_in[0];
  const float* V  = (const float*)d_in[1];
  const float* G  = (const float*)d_in[2];
  const float* S0 = (const float*)d_in[3];
  float* Out = (float*)d_out;

  const int B = in_sizes[0] / (T_DIM * H_DIM * K_DIM);   // = 4

  const size_t shmem =
      (size_t)(CH * K_DIM * 2 + K_DIM * CH + VL * CH + CH * CH + VL * K_DIM) *
          sizeof(unsigned short) +
      (size_t)(K_DIM * VL + CH * K_DIM + 2 * K_DIM) * sizeof(float);   // ~145 KB

  (void)hipFuncSetAttribute(reinterpret_cast<const void*>(gla_fwd_kernel),
                            hipFuncAttributeMaxDynamicSharedMemorySize, (int)shmem);

  dim3 grid(B * H_DIM * VSPLIT);   // 128 workgroups
  gla_fwd_kernel<<<grid, 256, shmem, stream>>>(Q, V, G, S0, Out);
}
